// CrossAttention_5239860101227
// MI455X (gfx1250) — matrix operand order repaired and checked by reading, not compiled, not run on hardware
//
#include <hip/hip_runtime.h>
#include <hip/hip_bf16.h>
#include <math.h>

// ---------------------------------------------------------------------------
// CDNA5 (gfx1250) cross-attention: bf16 WMMA + TDM (tensor_load_to_lds) GEMMs.
// ---------------------------------------------------------------------------

typedef __bf16 bf16;
typedef __attribute__((ext_vector_type(16))) __bf16 v16bf;
typedef __attribute__((ext_vector_type(8)))  __bf16 bf16x8;
typedef __attribute__((ext_vector_type(8)))  float  v8f;
typedef __attribute__((ext_vector_type(4)))  unsigned int u32x4;
typedef __attribute__((ext_vector_type(8)))  int i32x8;
typedef __attribute__((ext_vector_type(4)))  int i32x4;

#define B_  8
#define T_  1024
#define TE_ 256
#define C_  1024
#define CE_ 768
#define H_  16
#define HD_ 64

__device__ __forceinline__ v8f zero_v8f() {
  v8f z;
#pragma unroll
  for (int i = 0; i < 8; ++i) z[i] = 0.0f;
  return z;
}

// A-fragment (16x32 bf16, row-major source, ld = row stride in elements).
// Lanes 0-15: row M=lane, K = 0..15; lanes 16-31: row M=lane-16,
// K = 16..31 (matches load_b_frag K order).
__device__ __forceinline__ v16bf load_a_frag(const bf16* __restrict__ A, long ld,
                                             int row0, int k0, int lane) {
  int m    = lane & 15;
  int koff = (lane & 16) ? 16 : 0;
  const bf16* p = A + (long)(row0 + m) * ld + (k0 + koff);
  bf16x8 lo = *(const bf16x8*)(p);
  bf16x8 hi = *(const bf16x8*)(p + 8);
  v16bf a;
#pragma unroll
  for (int i = 0; i < 8; ++i) { a[i] = lo[i]; a[8 + i] = hi[i]; }
  return a;
}

// B-fragment (32x16 bf16) from a K-contiguous source (B^T stored [N,K]).
// Lanes 0-15: col N=lane, K=0..15; lanes 16-31: col N=lane-16, K=16..31.
__device__ __forceinline__ v16bf load_b_frag(const bf16* __restrict__ Bt, long ld,
                                             int col0, int k0, int lane) {
  int n  = lane & 15;
  int kb = (lane & 16) ? 16 : 0;
  const bf16* p = Bt + (long)(col0 + n) * ld + (k0 + kb);
  bf16x8 lo = *(const bf16x8*)(p);
  bf16x8 hi = *(const bf16x8*)(p + 8);
  v16bf b;
#pragma unroll
  for (int i = 0; i < 8; ++i) { b[i] = lo[i]; b[8 + i] = hi[i]; }
  return b;
}

// ---------------------------------------------------------------------------
// Tensor Data Mover: 2D tile (tile_h rows x tile_w bf16 elems) global -> LDS.
// Builds the D# per CDNA5 ISA ch.8: group0 {count=1, lds_addr, global_addr,
// type=2}, group1 {data_size=2B, tensor dims, tile dims, dim0 stride}.
// Issued per-wave (EXEC ignored); tracked by TENSORcnt.
// ---------------------------------------------------------------------------
__device__ __forceinline__ void tdm_load_2d(const bf16* gptr, unsigned lds_off,
                                            unsigned tile_w, unsigned tile_h,
                                            unsigned row_stride_elems) {
  unsigned long long ga = (unsigned long long)(size_t)gptr;
  u32x4 g0;
  g0[0] = 1u;                                        // count=1, no gather
  g0[1] = lds_off;                                   // LDS byte address
  g0[2] = (unsigned)(ga & 0xffffffffu);              // global_addr[31:0]
  g0[3] = (unsigned)((ga >> 32) & 0x1ffffffu) | (2u << 30);  // [56:32] | type=2

  unsigned td0 = row_stride_elems;   // tensor_dim0 >= tile_w (in-bounds tiles)
  unsigned td1 = tile_h;             // tensor_dim1
  unsigned long long st0 = row_stride_elems;  // tensor_dim0_stride (elems)
  i32x8 g1;
  g1[0] = (int)(1u << 16);                               // data_size=1 (2B)
  g1[1] = (int)((td0 & 0xffffu) << 16);                  // dim0[15:0]
  g1[2] = (int)(((td0 >> 16) & 0xffffu) | ((td1 & 0xffffu) << 16));
  g1[3] = (int)(((td1 >> 16) & 0xffffu) | ((tile_w & 0xffffu) << 16));
  g1[4] = (int)(tile_h & 0xffffu);                       // tile_dim1, tile_dim2=0
  g1[5] = (int)(unsigned)(st0 & 0xffffffffu);
  g1[6] = (int)(unsigned)((st0 >> 32) & 0xffffu);        // stride1=0
  g1[7] = 0;

  i32x4 gz = {0, 0, 0, 0};
#if defined(__clang_major__) && (__clang_major__ >= 23)
  i32x8 gz8 = {0, 0, 0, 0, 0, 0, 0, 0};
  __builtin_amdgcn_tensor_load_to_lds(g0, g1, gz, gz, gz8, 0);
#else
  __builtin_amdgcn_tensor_load_to_lds(g0, g1, gz, gz, 0);
#endif
}

// ---------------------------------------------------------------------------
// Elementwise helpers
// ---------------------------------------------------------------------------
__global__ void cvt_f32_bf16(const float* __restrict__ src, bf16* __restrict__ dst, long n) {
  long i = (long)blockIdx.x * blockDim.x + threadIdx.x;
  long stride = (long)gridDim.x * blockDim.x;
  for (; i < n; i += stride) dst[i] = (bf16)src[i];
}

// W [K,N] f32 (row-major) -> Wt [N,K] bf16
__global__ void transpose_f32_bf16(const float* __restrict__ W, bf16* __restrict__ Wt,
                                   int K, int N) {
  long i = (long)blockIdx.x * blockDim.x + threadIdx.x;
  long total = (long)K * N;
  long stride = (long)gridDim.x * blockDim.x;
  for (; i < total; i += stride) {
    int k = (int)(i / N);
    int n = (int)(i % N);
    Wt[(long)n * K + k] = (bf16)W[i];
  }
}

__global__ void zero_f32(float* __restrict__ p, long n) {
  long i = (long)blockIdx.x * blockDim.x + threadIdx.x;
  long stride = (long)gridDim.x * blockDim.x;
  for (; i < n; i += stride) p[i] = 0.0f;
}

// ---------------------------------------------------------------------------
// bf16 GEMM with TDM double-buffered LDS staging:
//   Out[M,N] = A[M,K] @ Bt[N,K]^T + bias[N]
// Block = 256 threads = 8 waves; block tile = 128 rows x 64 cols; K-step 64.
// Wave 0 drives the Tensor Data Mover (per-wave TENSORcnt), all waves compute
// from LDS. Grid: x = N/64, y = M/128. Requires K % 64 == 0, exact grid.
// ---------------------------------------------------------------------------
template <typename OutT>
__global__ __launch_bounds__(256) void gemm_bf16_tdm(const bf16* __restrict__ A,
                                                     const bf16* __restrict__ Bt,
                                                     const float* __restrict__ bias,
                                                     OutT* __restrict__ Out,
                                                     int M, int N, int K) {
  __shared__ bf16 sA[2][128][64];   // 32 KB
  __shared__ bf16 sB[2][64][64];    // 16 KB

  int lane = threadIdx.x & 31;
  int wave = threadIdx.x >> 5;
  int row_blk = blockIdx.y * 128;
  int col_blk = blockIdx.x * 64;

  const bf16* Ab = A + (long)row_blk * K;
  const bf16* Bb = Bt + (long)col_blk * K;

  // Prologue: stage K-step 0 into buffer 0.
  if (wave == 0) {
    tdm_load_2d(Ab, (unsigned)(size_t)&sA[0][0][0], 64, 128, (unsigned)K);
    tdm_load_2d(Bb, (unsigned)(size_t)&sB[0][0][0], 64, 64,  (unsigned)K);
  }

  v8f acc[4];
#pragma unroll
  for (int j = 0; j < 4; ++j) acc[j] = zero_v8f();

  const int nsteps = K >> 6;
  int buf = 0;
  for (int s = 0; s < nsteps; ++s) {
    if (wave == 0) {
      if (s + 1 < nsteps) {
        int k0 = (s + 1) << 6;
        // Issue next buffer first, then wait for current (same-wave TDM ops
        // complete in order, so tensorcnt<=2 means the first pair landed).
        tdm_load_2d(Ab + k0, (unsigned)(size_t)&sA[buf ^ 1][0][0], 64, 128, (unsigned)K);
        tdm_load_2d(Bb + k0, (unsigned)(size_t)&sB[buf ^ 1][0][0], 64, 64,  (unsigned)K);
        __builtin_amdgcn_s_wait_tensorcnt((short)2);
      } else {
        __builtin_amdgcn_s_wait_tensorcnt((short)0);
      }
    }
    __syncthreads();   // publish TDM-written LDS to all 8 waves

    const bf16* aT = &sA[buf][wave * 16][0];
    const bf16* bT = &sB[buf][0][0];
#pragma unroll
    for (int kk = 0; kk < 64; kk += 32) {
      v16bf a = load_a_frag(aT, 64, 0, kk, lane);
#pragma unroll
      for (int j = 0; j < 4; ++j) {
        v16bf b = load_b_frag(bT, 64, j * 16, kk, lane);
        acc[j] = __builtin_amdgcn_wmma_f32_16x16x32_bf16(false, a, false, b,
                                                         (short)0, acc[j], false, false);
      }
    }
    __syncthreads();   // all waves done with buf before it is re-filled
    buf ^= 1;
  }

  // D layout: lane%16 = col, VGPR r = row (lanes>=16: row+8)
  int col   = lane & 15;
  int rbase = (lane & 16) ? 8 : 0;
  int row0  = row_blk + wave * 16;
#pragma unroll
  for (int j = 0; j < 4; ++j) {
    int c = col_blk + 16 * j + col;
    float bv = bias[c];
#pragma unroll
    for (int r = 0; r < 8; ++r) {
      long rr = row0 + rbase + r;
      Out[rr * N + c] = (OutT)(acc[j][r] + bv);
    }
  }
}

// ---------------------------------------------------------------------------
// Fused attention per (b, h, 16-row tile of T).
// Block = 64 threads = 2 waves; each wave owns one 16-row tile.
// grid.x = B*H, grid.y = T / 32.
// ---------------------------------------------------------------------------
__global__ __launch_bounds__(64) void attn_bf16_wmma(const bf16* __restrict__ qb,
                                                     const bf16* __restrict__ kb,
                                                     const bf16* __restrict__ vb,
                                                     bf16* __restrict__ yb,
                                                     float* __restrict__ att_mean) {
  __shared__ float sS[2][16][TE_];   // f32 scores / exp
  __shared__ bf16  sP[2][16][TE_];   // bf16 probabilities

  int lane = threadIdx.x & 31;
  int wave = threadIdx.x >> 5;
  int b = blockIdx.x / H_;
  int h = blockIdx.x % H_;
  int t0 = blockIdx.y * 32 + wave * 16;

  const bf16* qptr = qb + (long)b * T_ * C_ + h * HD_;
  const bf16* kptr = kb + (long)b * TE_ * C_ + h * HD_;
  const bf16* vptr = vb + (long)b * TE_ * C_ + h * HD_;

  // ---- scores S = (q @ k^T) * 1/sqrt(HD) -------------------------------
  v16bf aq0 = load_a_frag(qptr, C_, t0, 0,  lane);
  v16bf aq1 = load_a_frag(qptr, C_, t0, 32, lane);
  int col   = lane & 15;
  int rbase = (lane & 16) ? 8 : 0;

#pragma unroll
  for (int tt = 0; tt < 16; ++tt) {
    v8f acc = zero_v8f();
    v16bf bk0 = load_b_frag(kptr, C_, tt * 16, 0,  lane);
    acc = __builtin_amdgcn_wmma_f32_16x16x32_bf16(false, aq0, false, bk0, (short)0, acc, false, false);
    v16bf bk1 = load_b_frag(kptr, C_, tt * 16, 32, lane);
    acc = __builtin_amdgcn_wmma_f32_16x16x32_bf16(false, aq1, false, bk1, (short)0, acc, false, false);
#pragma unroll
    for (int r = 0; r < 8; ++r)
      sS[wave][rbase + r][tt * 16 + col] = acc[r] * 0.125f;  // 1/sqrt(64)
  }
  __syncthreads();

  // ---- softmax over TE=256, one row per lane pair (lane, lane^16) ------
  int row  = lane & 15;
  int half = lane >> 4;
  float* srow = &sS[wave][row][half * 128];

  float mx = -1e30f;
#pragma unroll 8
  for (int j = 0; j < 128; ++j) mx = fmaxf(mx, srow[j]);
  mx = fmaxf(mx, __shfl_xor(mx, 16, 32));

  float sum = 0.0f;
#pragma unroll 8
  for (int j = 0; j < 128; ++j) {
    float e = __expf(srow[j] - mx);
    srow[j] = e;
    sum += e;
  }
  sum += __shfl_xor(sum, 16, 32);
  float inv = 1.0f / sum;

  float* amrow = att_mean + ((long)b * T_ + t0 + row) * TE_ + half * 128;
#pragma unroll 4
  for (int j = 0; j < 128; ++j) {
    float p = srow[j] * inv;
    sP[wave][row][half * 128 + j] = (bf16)p;
    atomicAdd(&amrow[j], p * (1.0f / (float)H_));   // mean over heads
  }
  __syncthreads();

  // ---- y = P @ V  (16x256 @ 256x64) ------------------------------------
#pragma unroll
  for (int dt = 0; dt < 4; ++dt) {
    v8f acc = zero_v8f();
    for (int k0 = 0; k0 < TE_; k0 += 32) {
      v16bf ap = load_a_frag(&sP[wave][0][0], TE_, 0, k0, lane);
      v16bf bv;
      {
        int n   = lane & 15;
        int kb2 = (lane & 16) ? 16 : 0;
        const bf16* p = vptr + (long)(k0 + kb2) * C_ + dt * 16 + n;
#pragma unroll
        for (int j = 0; j < 16; ++j) bv[j] = p[(long)j * C_];
      }
      acc = __builtin_amdgcn_wmma_f32_16x16x32_bf16(false, ap, false, bv, (short)0, acc, false, false);
    }
    bf16* yrow = yb + ((long)b * T_ + t0) * C_ + h * HD_ + dt * 16;
#pragma unroll
    for (int r = 0; r < 8; ++r)
      yrow[(long)(rbase + r) * C_ + col] = (bf16)acc[r];
  }
}

// ---------------------------------------------------------------------------
// Host-side launch
// ---------------------------------------------------------------------------
extern "C" void kernel_launch(void* const* d_in, const int* in_sizes, int n_in,
                              void* d_out, int out_size, void* d_ws, size_t ws_size,
                              hipStream_t stream) {
  (void)in_sizes; (void)n_in; (void)out_size; (void)ws_size;

  const float* x   = (const float*)d_in[0];
  const float* enc = (const float*)d_in[1];
  const float* Wq  = (const float*)d_in[2];
  const float* bq  = (const float*)d_in[3];
  const float* Wk  = (const float*)d_in[4];
  const float* bk  = (const float*)d_in[5];
  const float* Wv  = (const float*)d_in[6];
  const float* bv  = (const float*)d_in[7];
  const float* Wp  = (const float*)d_in[8];
  const float* bp  = (const float*)d_in[9];

  float* y_out  = (float*)d_out;                        // [B,T,C]
  float* am_out = (float*)d_out + (long)B_ * T_ * C_;   // [B,T,TE]

  char* ws = (char*)d_ws;
  size_t off = 0;
  auto carve = [&](size_t bytes) {
    size_t o = off;
    off += (bytes + 255) & ~(size_t)255;
    return o;
  };
  bf16* xb  = (bf16*)(ws + carve((size_t)B_ * T_ * C_ * 2));
  bf16* eb  = (bf16*)(ws + carve((size_t)B_ * TE_ * CE_ * 2));
  bf16* Wqt = (bf16*)(ws + carve((size_t)C_ * C_ * 2));
  bf16* Wkt = (bf16*)(ws + carve((size_t)CE_ * C_ * 2));
  bf16* Wvt = (bf16*)(ws + carve((size_t)CE_ * C_ * 2));
  bf16* Wpt = (bf16*)(ws + carve((size_t)C_ * C_ * 2));
  bf16* qb  = (bf16*)(ws + carve((size_t)B_ * T_ * C_ * 2));
  bf16* kbuf= (bf16*)(ws + carve((size_t)B_ * TE_ * C_ * 2));
  bf16* vbuf= (bf16*)(ws + carve((size_t)B_ * TE_ * C_ * 2));
  bf16* yb  = (bf16*)(ws + carve((size_t)B_ * T_ * C_ * 2));

  // 1) precision conversion + weight transposes
  {
    long n = (long)B_ * T_ * C_;
    cvt_f32_bf16<<<4096, 256, 0, stream>>>(x, xb, n);
    n = (long)B_ * TE_ * CE_;
    cvt_f32_bf16<<<2048, 256, 0, stream>>>(enc, eb, n);
    transpose_f32_bf16<<<2048, 256, 0, stream>>>(Wq, Wqt, C_, C_);
    transpose_f32_bf16<<<2048, 256, 0, stream>>>(Wk, Wkt, CE_, C_);
    transpose_f32_bf16<<<2048, 256, 0, stream>>>(Wv, Wvt, CE_, C_);
    transpose_f32_bf16<<<2048, 256, 0, stream>>>(Wp, Wpt, C_, C_);
  }

  // 2) projections (TDM-staged WMMA GEMMs)
  {
    dim3 blk(256);
    dim3 gq(C_ / 64, (B_ * T_) / 128);     // 16 x 64
    gemm_bf16_tdm<bf16><<<gq, blk, 0, stream>>>(xb, Wqt, bq, qb, B_ * T_, C_, C_);
    dim3 gkv(C_ / 64, (B_ * TE_) / 128);   // 16 x 16
    gemm_bf16_tdm<bf16><<<gkv, blk, 0, stream>>>(eb, Wkt, bk, kbuf, B_ * TE_, C_, CE_);
    gemm_bf16_tdm<bf16><<<gkv, blk, 0, stream>>>(eb, Wvt, bv, vbuf, B_ * TE_, C_, CE_);
  }

  // 3) attention (fused scores + softmax + PV, accumulates att_mean)
  zero_f32<<<2048, 256, 0, stream>>>(am_out, (long)B_ * T_ * TE_);
  {
    dim3 blk(64);
    dim3 grd(B_ * H_, T_ / 32);
    attn_bf16_wmma<<<grd, blk, 0, stream>>>(qb, kbuf, vbuf, yb, am_out);
  }

  // 4) output projection (TDM-staged WMMA GEMM, f32 out + bias)
  {
    dim3 blk(256);
    dim3 g(C_ / 64, (B_ * T_) / 128);
    gemm_bf16_tdm<float><<<g, blk, 0, stream>>>(yb, Wpt, bp, y_out, B_ * T_, C_, C_);
  }
}
